// GraphAttentionFraudDetector_82068235092028
// MI455X (gfx1250) — matrix operand order repaired and checked by reading, not compiled, not run on hardware
//
#include <hip/hip_runtime.h>
#include <hip/hip_bf16.h>
#include <math.h>

typedef __attribute__((ext_vector_type(16))) __bf16 v16bf;
typedef __attribute__((ext_vector_type(8)))  __bf16 v8bf;
typedef __attribute__((ext_vector_type(2)))  __bf16 v2bf;
typedef __attribute__((ext_vector_type(8)))  float  v8f;
typedef __attribute__((ext_vector_type(4)))  unsigned int u32x4;
typedef __attribute__((ext_vector_type(8)))  int i32x8;
typedef __attribute__((ext_vector_type(4)))  int i32x4;

#define LRELU_SLOPE 0.2f
#define BN_EPS 1e-5f

#if __has_builtin(__builtin_amdgcn_tensor_load_to_lds)
#define HAVE_TDM 1
#endif

// ---------------------------------------------------------------------------
// f32 -> bf16 bulk convert (pairs -> v_cvt_pk_bf16_f32)
// ---------------------------------------------------------------------------
__global__ void cvt_f32_bf16_kernel(const float* __restrict__ in,
                                    __bf16* __restrict__ out, int npairs) {
  int idx = blockIdx.x * blockDim.x + threadIdx.x;
  if (idx >= npairs) return;
  float2 v = ((const float2*)in)[idx];
  v2bf p = {(__bf16)v.x, (__bf16)v.y};
  ((v2bf*)out)[idx] = p;
}

// ---------------------------------------------------------------------------
// Pack W[K,Ncol] (f32) into WMMA B-fragment order, bf16:
//   out[(((ng*ksteps + ks)*4 + j)*32 + lane)*16 + i] = W[k*Ncol + col]
//   k = ks*32 + (lane>>4)*16 + i ; col = ng*64 + j*16 + (lane&15)
// ---------------------------------------------------------------------------
__global__ void pack_b_kernel(const float* __restrict__ W, __bf16* __restrict__ out,
                              int K, int Ncol) {
  int idx = blockIdx.x * blockDim.x + threadIdx.x;
  if (idx >= K * Ncol) return;
  const int ksteps = K >> 5;
  int i    = idx & 15;
  int lane = (idx >> 4) & 31;
  int j    = (idx >> 9) & 3;
  int rest = idx >> 11;
  int ks   = rest % ksteps;
  int ng   = rest / ksteps;
  int k    = ks * 32 + ((lane >> 4) << 4) + i;
  int col  = ng * 64 + j * 16 + (lane & 15);
  out[idx] = (__bf16)W[(size_t)k * Ncol + col];
}

// ---------------------------------------------------------------------------
// bf16 WMMA GEMM: C[M,Ncol] = A[M,K] * B[K,Ncol], fp32 accumulate.
// 8 waves/block, each wave a 16x64 strip; all waves share one 64-column
// packed-B slab staged in LDS by the Tensor Data Mover (fallback: coop copy).
// ---------------------------------------------------------------------------
__global__ void __launch_bounds__(256) wmma_gemm_kernel(
    const __bf16* __restrict__ Abf, const __bf16* __restrict__ Bpack,
    float* __restrict__ C, int M, int K, int Ncol) {
  __shared__ v16bf sB[1024];                 // up to 32 KB of staged weights
  const int wave = threadIdx.x >> 5;
  const int lane = threadIdx.x & 31;
  const int NtG = Ncol >> 6;
  const int Mt = M >> 4;
  const int ksteps = K >> 5;
  const int ng = blockIdx.x % NtG;
  const int mb = blockIdx.x / NtG;
  const int slabBytes = ksteps << 12;        // ksteps * 4 frags * 32 lanes * 32B

  const __bf16* slab = Bpack + (size_t)ng * (size_t)(slabBytes >> 1);
#ifdef HAVE_TDM
  if (threadIdx.x < 32) {                    // wave 0 issues the TDM descriptor
    unsigned long long ga = (unsigned long long)(const void*)slab;
    unsigned int lds_off  = (unsigned int)(unsigned long long)(void*)&sB[0];
    unsigned int Ldw = (unsigned int)(slabBytes >> 2);   // dwords, <= 8192
    u32x4 g0 = {1u,                                       // count=1 valid descriptor
                lds_off,                                  // lds_addr
                (unsigned int)(ga & 0xFFFFFFFFu),         // global_addr[31:0]
                (unsigned int)((ga >> 32) & 0x01FFFFFFu) | (2u << 30)}; // addr hi | type=2
    i32x8 g1 = {(int)0x00020000,                          // data_size = 4B
                (int)((Ldw & 0xFFFFu) << 16),             // tensor_dim0[15:0]
                (int)((Ldw >> 16) | (1u << 16)),          // tensor_dim0[31:16] | tensor_dim1=1
                (int)((Ldw & 0xFFFFu) << 16),             // tile_dim0 = Ldw
                0,                                        // tile_dim1/2 unused
                (int)Ldw,                                 // tensor_dim0_stride
                0, 0};
    i32x4 gz = {0, 0, 0, 0};
#if __clang_major__ >= 23
    i32x8 z8 = {0, 0, 0, 0, 0, 0, 0, 0};
    __builtin_amdgcn_tensor_load_to_lds(g0, g1, gz, gz, z8, 0);
#else
    __builtin_amdgcn_tensor_load_to_lds(g0, g1, gz, gz, 0);
#endif
#if __has_builtin(__builtin_amdgcn_s_wait_tensorcnt)
    __builtin_amdgcn_s_wait_tensorcnt(0);
#endif
  }
#else
  {
    const uint4* s4 = (const uint4*)slab;
    uint4* d4 = (uint4*)sB;
    for (int i = threadIdx.x; i < (slabBytes >> 4); i += 256) d4[i] = s4[i];
  }
#endif
  __syncthreads();

  int mt = mb * 8 + wave;
  const bool active = (mt < Mt);
  if (!active) mt = Mt - 1;                  // clamp; keep wave alive, skip store
  const int tm = mt << 4;
  const int tn = ng << 6;
  const int half = lane >> 4;
  const int l15 = lane & 15;
  const __bf16* __restrict__ arow = Abf + (size_t)(tm + l15) * K;

  v8f z = {0.f, 0.f, 0.f, 0.f, 0.f, 0.f, 0.f, 0.f};
  v8f acc[4] = {z, z, z, z};

  for (int ks = 0; ks < ksteps; ++ks) {
    // A fragment (contiguous K, matches packed B): K = ka..ka+15
    const v8bf* ap = (const v8bf*)(arow + ks * 32 + half * 16);
    v8bf lo = ap[0];
    v8bf hi = ap[1];
    v16bf av = __builtin_shufflevector(lo, hi, 0, 1, 2, 3, 4, 5, 6, 7,
                                       8, 9, 10, 11, 12, 13, 14, 15);
    const v16bf* bf = &sB[(ks << 7) + lane];
#pragma unroll
    for (int j = 0; j < 4; ++j) {
      v16bf bv = bf[j * 32];                 // ds_load of pre-swizzled fragment
      acc[j] = __builtin_amdgcn_wmma_f32_16x16x32_bf16(
          false, av, false, bv, (short)0, acc[j], false, false);
    }
  }

  if (active) {
    const int mbase = tm + half * 8;         // C/D: lanes 0-15 -> M=r, 16-31 -> M=r+8
    float* __restrict__ cp = C + (size_t)mbase * Ncol + tn + l15;
#pragma unroll
    for (int j = 0; j < 4; ++j) {
      float* cpj = cp + j * 16;
#pragma unroll
      for (int r = 0; r < 8; ++r) { *cpj = acc[j][r]; cpj += Ncol; }
    }
  }
}

// ---------------------------------------------------------------------------
__global__ void attn_logit_kernel(const float* __restrict__ h,
                                  const float* __restrict__ a_s,
                                  const float* __restrict__ a_d,
                                  float* __restrict__ als, float* __restrict__ ald,
                                  int N, int H, int C) {
  int idx = blockIdx.x * blockDim.x + threadIdx.x;
  if (idx >= N * H) return;
  int n = idx / H, hd = idx % H;
  const float* row = h + (size_t)n * H * C + (size_t)hd * C;
  const float* vs = a_s + (size_t)hd * C;
  const float* vd = a_d + (size_t)hd * C;
  float ss = 0.f, sd = 0.f;
  for (int c = 0; c < C; ++c) { float x = row[c]; ss += x * vs[c]; sd += x * vd[c]; }
  als[idx] = ss; ald[idx] = sd;
}

__global__ void fill_kernel(float* __restrict__ p, float v, int n) {
  int idx = blockIdx.x * blockDim.x + threadIdx.x;
  if (idx < n) p[idx] = v;
}

__device__ __forceinline__ void atomic_max_float(float* addr, float v) {
  if (v >= 0.0f) atomicMax((int*)addr, __float_as_int(v));
  else           atomicMin((unsigned int*)addr, __float_as_uint(v));
}

__global__ void edge_max_kernel(const int* __restrict__ src, const int* __restrict__ dst,
                                int E, int N, const float* __restrict__ als,
                                const float* __restrict__ ald, float* __restrict__ m, int H) {
  int idx = blockIdx.x * blockDim.x + threadIdx.x;
  int tot = E + N;
  if (idx >= tot) return;
  int s, d;
  if (idx < E) { s = src[idx]; d = dst[idx]; } else { s = idx - E; d = s; }
  for (int h = 0; h < H; ++h) {
    float e = als[(size_t)s * H + h] + ald[(size_t)d * H + h];
    e = (e > 0.0f) ? e : LRELU_SLOPE * e;
    atomic_max_float(&m[(size_t)d * H + h], e);
  }
}

__global__ void __launch_bounds__(64) edge_accum_kernel(
    const int* __restrict__ src, const int* __restrict__ dst, int E, int N,
    const float* __restrict__ als, const float* __restrict__ ald,
    const float* __restrict__ m, float* __restrict__ denom,
    const float* __restrict__ hfeat, float* __restrict__ acc, int H, int C) {
  int eidx = blockIdx.x;
  int s, d;
  if (eidx < E) { s = src[eidx]; d = dst[eidx]; } else { s = eidx - E; d = s; }
  const int HC = H * C;
  __builtin_prefetch(&hfeat[(size_t)s * HC], 0, 0);   // global_prefetch_b8
  if ((int)threadIdx.x < H) {
    int hd = threadIdx.x;
    float e = als[(size_t)s * H + hd] + ald[(size_t)d * H + hd];
    e = (e > 0.0f) ? e : LRELU_SLOPE * e;
    float ee = expf(e - m[(size_t)d * H + hd]);
    atomicAdd(&denom[(size_t)d * H + hd], ee);
  }
  const int V = HC / blockDim.x;
  for (int v = 0; v < V; ++v) {
    int c = (int)threadIdx.x * V + v;
    int hd = c / C;
    float e = als[(size_t)s * H + hd] + ald[(size_t)d * H + hd];
    e = (e > 0.0f) ? e : LRELU_SLOPE * e;
    float ee = expf(e - m[(size_t)d * H + hd]);
    atomicAdd(&acc[(size_t)d * HC + c], ee * hfeat[(size_t)s * HC + c]);
  }
}

__global__ void finalize_kernel(float* __restrict__ acc, const float* __restrict__ denom,
                                const float* __restrict__ bias, int N, int H, int C, int elu) {
  int idx = blockIdx.x * blockDim.x + threadIdx.x;
  const int HC = H * C;
  if (idx >= N * HC) return;
  int col = idx % HC;
  int n = idx / HC;
  int hd = col / C;
  float v = acc[idx] / denom[(size_t)n * H + hd] + bias[col];
  if (elu) v = (v > 0.0f) ? v : expm1f(v);
  acc[idx] = v;
}

__global__ void bn_stats_kernel(const float* __restrict__ X, int Nrows, int HC,
                                int rowsPerBlock, float* __restrict__ sum,
                                float* __restrict__ sq) {
  int col = threadIdx.x;
  int r0 = blockIdx.x * rowsPerBlock;
  float s = 0.f, q = 0.f;
  for (int r = 0; r < rowsPerBlock; ++r) {
    int row = r0 + r;
    if (row < Nrows) { float v = X[(size_t)row * HC + col]; s += v; q += v * v; }
  }
  atomicAdd(&sum[col], s);
  atomicAdd(&sq[col], q);
}

__global__ void bn_apply_elu_kernel(float* __restrict__ X, const float* __restrict__ sum,
                                    const float* __restrict__ sq, const float* __restrict__ g,
                                    const float* __restrict__ be, int Nrows, int HC) {
  int idx = blockIdx.x * blockDim.x + threadIdx.x;
  if (idx >= Nrows * HC) return;
  int col = idx % HC;
  float inv_n = 1.0f / (float)Nrows;
  float mu = sum[col] * inv_n;
  float var = sq[col] * inv_n - mu * mu;
  float v = g[col] * (X[idx] - mu) * rsqrtf(var + BN_EPS) + be[col];
  X[idx] = (v > 0.0f) ? v : expm1f(v);
}

__global__ void __launch_bounds__(256) mlp_head_kernel(
    const float* __restrict__ in, const float* __restrict__ fw1,
    const float* __restrict__ fb1, const float* __restrict__ fw2,
    const float* __restrict__ fb2, float* __restrict__ out,
    int N, int Cin, int Chid, int Cout) {
  __shared__ float sW1[2048];
  __shared__ float sB1[64];
  __shared__ float sW2[128];
  __shared__ float sB2[8];
  for (int i = threadIdx.x; i < Cin * Chid; i += blockDim.x) sW1[i] = fw1[i];
  for (int i = threadIdx.x; i < Chid;       i += blockDim.x) sB1[i] = fb1[i];
  for (int i = threadIdx.x; i < Chid * Cout; i += blockDim.x) sW2[i] = fw2[i];
  for (int i = threadIdx.x; i < Cout;       i += blockDim.x) sB2[i] = fb2[i];
  __syncthreads();
  int n = blockIdx.x * blockDim.x + threadIdx.x;
  if (n >= N) return;
  float xin[64];
  for (int c = 0; c < Cin; ++c) xin[c] = in[(size_t)n * Cin + c];
  float o0 = sB2[0];
  float o1 = (Cout > 1) ? sB2[1] : 0.0f;
  for (int j = 0; j < Chid; ++j) {
    float a = sB1[j];
    for (int c = 0; c < Cin; ++c) a += xin[c] * sW1[c * Chid + j];
    a = fmaxf(a, 0.0f);
    o0 += a * sW2[j * Cout + 0];
    if (Cout > 1) o1 += a * sW2[j * Cout + 1];
  }
  out[(size_t)n * Cout + 0] = o0;
  if (Cout > 1) out[(size_t)n * Cout + 1] = o1;
}

// ---------------------------------------------------------------------------
static inline int cdiv(int a, int b) { return (a + b - 1) / b; }

static void launch_gat_layer(const float* in, int Fin, const float* W,
                             const float* a_s, const float* a_d, const float* bias,
                             int H, int C, int N, int E,
                             const int* srcp, const int* dstp,
                             float* hbuf, float* accbuf, float* als, float* ald,
                             float* mbuf, float* dbuf, __bf16* abf, __bf16* wpack,
                             int elu_flag, hipStream_t stream) {
  const int HC = H * C;
  cvt_f32_bf16_kernel<<<cdiv(N * Fin / 2, 256), 256, 0, stream>>>(in, abf, N * Fin / 2);
  pack_b_kernel<<<cdiv(Fin * HC, 256), 256, 0, stream>>>(W, wpack, Fin, HC);
  const int nblocks = cdiv(N / 16, 8) * (HC / 64);
  wmma_gemm_kernel<<<nblocks, 256, 0, stream>>>(abf, wpack, hbuf, N, Fin, HC);
  attn_logit_kernel<<<cdiv(N * H, 256), 256, 0, stream>>>(hbuf, a_s, a_d, als, ald, N, H, C);
  fill_kernel<<<cdiv(N * H, 256), 256, 0, stream>>>(mbuf, -__builtin_huge_valf(), N * H);
  fill_kernel<<<cdiv(N * H, 256), 256, 0, stream>>>(dbuf, 0.0f, N * H);
  fill_kernel<<<cdiv(N * HC, 256), 256, 0, stream>>>(accbuf, 0.0f, N * HC);
  const int tot = E + N;
  edge_max_kernel<<<cdiv(tot, 256), 256, 0, stream>>>(srcp, dstp, E, N, als, ald, mbuf, H);
  edge_accum_kernel<<<tot, 64, 0, stream>>>(srcp, dstp, E, N, als, ald, mbuf, dbuf,
                                            hbuf, accbuf, H, C);
  finalize_kernel<<<cdiv(N * HC, 256), 256, 0, stream>>>(accbuf, dbuf, bias, N, H, C, elu_flag);
}

extern "C" void kernel_launch(void* const* d_in, const int* in_sizes, int n_in,
                              void* d_out, int out_size, void* d_ws, size_t ws_size,
                              hipStream_t stream) {
  (void)n_in; (void)out_size; (void)ws_size;
  const float* x   = (const float*)d_in[0];
  const int*   ei  = (const int*)d_in[1];
  const float* W1  = (const float*)d_in[2];
  const float* a1s = (const float*)d_in[3];
  const float* a1d = (const float*)d_in[4];
  const float* b1  = (const float*)d_in[5];
  const float* W2  = (const float*)d_in[6];
  const float* a2s = (const float*)d_in[7];
  const float* a2d = (const float*)d_in[8];
  const float* b2  = (const float*)d_in[9];
  const float* W3  = (const float*)d_in[10];
  const float* a3s = (const float*)d_in[11];
  const float* a3d = (const float*)d_in[12];
  const float* b3  = (const float*)d_in[13];
  const float* g1  = (const float*)d_in[14];
  const float* be1 = (const float*)d_in[15];
  const float* g2  = (const float*)d_in[16];
  const float* be2 = (const float*)d_in[17];
  const float* fw1 = (const float*)d_in[18];
  const float* fb1 = (const float*)d_in[19];
  const float* fw2 = (const float*)d_in[20];
  const float* fb2 = (const float*)d_in[21];

  const int HC   = in_sizes[5];            // 256
  const int C3   = in_sizes[13];           // 64
  const int H    = in_sizes[3] / C3;       // 4
  const int C    = HC / H;                 // 64
  const int F    = in_sizes[2] / HC;       // 128
  const int N    = in_sizes[0] / F;        // 50000
  const int E    = in_sizes[1] / 2;        // 400000
  const int Chid = in_sizes[19];           // 32
  const int Cout = in_sizes[21];           // 2

  const int* srcp = ei;
  const int* dstp = ei + E;

  char* base = (char*)d_ws;
  float* bufA  = (float*)base;  base += (size_t)N * HC * 4;   // GEMM output h
  float* bufB  = (float*)base;  base += (size_t)N * HC * 4;   // aggregation / layer output
  float* als   = (float*)base;  base += (size_t)N * H * 4;
  float* ald   = (float*)base;  base += (size_t)N * H * 4;
  float* mbuf  = (float*)base;  base += (size_t)N * H * 4;
  float* dbuf  = (float*)base;  base += (size_t)N * H * 4;
  float* bnsum = (float*)base;  base += (size_t)HC * 4;
  float* bnsq  = (float*)base;  base += (size_t)HC * 4;
  __bf16* abf  = (__bf16*)base; base += (size_t)N * HC * 2;   // bf16 A (K<=HC)
  __bf16* wpk  = (__bf16*)base; base += (size_t)HC * HC * 2;  // packed bf16 W

  // ---- Layer 1: x[N,F] -> bufB[N,HC] (GAT), then BN+ELU in place ----
  launch_gat_layer(x, F, W1, a1s, a1d, b1, H, C, N, E, srcp, dstp,
                   bufA, bufB, als, ald, mbuf, dbuf, abf, wpk, /*elu=*/0, stream);
  fill_kernel<<<1, 256, 0, stream>>>(bnsum, 0.0f, HC);
  fill_kernel<<<1, 256, 0, stream>>>(bnsq, 0.0f, HC);
  bn_stats_kernel<<<cdiv(N, 256), HC, 0, stream>>>(bufB, N, HC, 256, bnsum, bnsq);
  bn_apply_elu_kernel<<<cdiv(N * HC, 256), 256, 0, stream>>>(bufB, bnsum, bnsq, g1, be1, N, HC);

  // ---- Layer 2 ----
  launch_gat_layer(bufB, HC, W2, a2s, a2d, b2, H, C, N, E, srcp, dstp,
                   bufA, bufB, als, ald, mbuf, dbuf, abf, wpk, /*elu=*/0, stream);
  fill_kernel<<<1, 256, 0, stream>>>(bnsum, 0.0f, HC);
  fill_kernel<<<1, 256, 0, stream>>>(bnsq, 0.0f, HC);
  bn_stats_kernel<<<cdiv(N, 256), HC, 0, stream>>>(bufB, N, HC, 256, bnsum, bnsq);
  bn_apply_elu_kernel<<<cdiv(N * HC, 256), 256, 0, stream>>>(bufB, bnsum, bnsq, g2, be2, N, HC);

  // ---- Layer 3 (single head, ELU fused) ----
  launch_gat_layer(bufB, HC, W3, a3s, a3d, b3, /*H=*/1, C3, N, E, srcp, dstp,
                   bufA, bufB, als, ald, mbuf, dbuf, abf, wpk, /*elu=*/1, stream);

  // ---- MLP head ----
  mlp_head_kernel<<<cdiv(N, 256), 256, 0, stream>>>(bufB, fw1, fb1, fw2, fb2,
                                                    (float*)d_out, N, C3, Chid, Cout);
}
